// SEDD4REC_61323543052350
// MI455X (gfx1250) — compile-verified, matrix operand order checked by reading, not run on hardware
//
#include <hip/hip_runtime.h>
#include <hip/hip_bf16.h>
#include <math.h>

// ---------------------------------------------------------------------------
// SEDD-style transformer forward for MI455X (gfx1250, wave32, WMMA).
// GEMM hot path: bf16 operands, f32 accumulation (v_wmma_f32_16x16x32_bf16).
// Tile staging: A via global_load_async_to_lds_b128 (ASYNCcnt),
//               W via TDM tensor_load_to_lds with a hand-built D# (TENSORcnt),
//               TDM pad_enable generates the LDS bank-conflict padding.
// fp32->bf16 conversion lives in the bandwidth-bound producer kernels only.
// ---------------------------------------------------------------------------

#define B_  256
#define T_  200
#define H_  512
#define NH_ 8
#define L_  2
#define CD_ 128
#define FE_ 256
#define DH_ 64            // head dim
#define HH_ (H_ * H_)     // 262144, one weight matrix

typedef __bf16 bf16_t;
typedef __attribute__((ext_vector_type(16))) __bf16 v16bf;
typedef __attribute__((ext_vector_type(8)))  float  v8f;
typedef __attribute__((ext_vector_type(4)))  unsigned v4u;
typedef __attribute__((ext_vector_type(8)))  unsigned v8u;

// fp32 -> bf16 round-to-nearest-even
__device__ __forceinline__ bf16_t f2bf(float x) {
    union { float f; unsigned u; } a; a.f = x;
    unsigned r = a.u + 0x7FFFu + ((a.u >> 16) & 1u);
    union { unsigned short s; bf16_t b; } o; o.s = (unsigned short)(r >> 16);
    return o.b;
}

// Load a 16x32 bf16 WMMA fragment from LDS (K contiguous per row).
// CDNA5 16-bit A/B layout: lanes 0-15 hold K 0..7 / 16..23,
// lanes 16-31 hold K 8..15 / 24..31 (two 16B chunks per lane).
__device__ __forceinline__ v16bf ld_frag(const bf16_t* p) {
    union { float4 f[2]; v16bf v; } u;
    u.f[0] = *(const float4*)(p);
    u.f[1] = *(const float4*)(p + 16);
    return u.v;
}

// Per-lane async 16B global -> LDS copy (tracked by ASYNCcnt).
__device__ __forceinline__ void async_ld_b128(unsigned lds_byte_addr, const void* g) {
    unsigned long long ga = (unsigned long long)(uintptr_t)g;
    asm volatile("global_load_async_to_lds_b128 %0, %1, off"
                 :: "v"(lds_byte_addr), "v"(ga) : "memory");
}

// TDM: load a 64x64 bf16 tile from a 512x512 bf16 row-major tensor into LDS,
// with 16B of LDS padding inserted after every 128B row (-> row pitch 144B).
// D# group0: count=1, lds_addr, 57b global_addr, type=2 ("image").
// D# group1: data_size=2B, pad_enable, pad_interval=32 dw, pad_amount=4 dw,
//            tensor_dim0/1=512, tile_dim0/1=64, stride0=512, stride1=512*512.
__device__ __forceinline__ void tdm_load_tile(unsigned lds_byte_addr, const void* gsrc) {
    unsigned long long ga = (unsigned long long)(uintptr_t)gsrc;
    v4u g0;
    g0[0] = 1u;                                   // count=1 (valid user D#)
    g0[1] = lds_byte_addr;                        // lds_addr [63:32]
    g0[2] = (unsigned)ga;                         // global_addr lo
    g0[3] = (unsigned)(ga >> 32) | 0x80000000u;   // global_addr hi | type=2
    v8u g1;
    g1[0] = 0x07110000u;   // data_size=1(2B)<<16 | pad_en<<20 | intv=4<<22 | amt=3<<25
    g1[1] = (512u & 0xFFFFu) << 16;               // tensor_dim0[15:0] in [63:48]
    g1[2] = (512u >> 16) | ((512u & 0xFFFFu) << 16); // dim0 hi | dim1 lo
    g1[3] = (512u >> 16) | (64u << 16);           // dim1 hi | tile_dim0=64
    g1[4] = 64u;                                  // tile_dim1=64 | tile_dim2=0
    g1[5] = 512u;                                 // tensor_dim0_stride lo32
    g1[6] = (262144u & 0xFFFFu) << 16;            // stride0 hi16 | stride1 lo16
    g1[7] = 262144u >> 16;                        // stride1[47:16]
    asm volatile("tensor_load_to_lds %0, %1" :: "s"(g0), "s"(g1) : "memory");
}

// ---------------------------------------------------------------------------
// WMMA GEMM: C[M,N] = act( A[M,K]bf16 @ Wt[N,K]bf16^T + bias + rowBias )
// Block: 256 threads (8 waves). Block tile 128x64, wave tile 32x32.
// ---------------------------------------------------------------------------
#define BM 128
#define BN 64
#define BK 64
#define LDA 72   // halves: 144B rows -> conflict-free fragment loads
#define LDB 72

__global__ __launch_bounds__(256)
void gemm_bf16_wmma(const bf16_t* __restrict__ A, const bf16_t* __restrict__ Wt,
                    const float* __restrict__ bias, const float* __restrict__ rowBias,
                    float* __restrict__ Cf, bf16_t* __restrict__ Cb,
                    int M, int N, int K, int rowDiv, int doRelu)
{
    __shared__ __align__(16) bf16_t As[BM * LDA];
    __shared__ __align__(16) bf16_t Bs[BN * LDB];

    const int tid  = threadIdx.x;
    const int lane = tid & 31;
    const int w    = tid >> 5;      // wave 0..7
    const int wm   = w >> 1;        // 0..3  (M direction)
    const int wn   = w & 1;         // 0..1  (N direction)
    const int mBlk = blockIdx.x * BM;
    const int nBlk = blockIdx.y * BN;

    const unsigned asBase = (unsigned)(uintptr_t)&As[0];
    const unsigned bsBase = (unsigned)(uintptr_t)&Bs[0];

    v8f acc[2][2] = {};
    const int l15  = lane & 15;
    const int hsel = (lane >> 4) * 8;   // 0 or 8 halves (16B)

    for (int kb = 0; kb < K; kb += BK) {
        // ---- W tile via TDM (one descriptor, wave 0 issues) ----
        if (w == 0)
            tdm_load_tile(bsBase, Wt + (size_t)nBlk * K + kb);
        // ---- A tile via per-lane async b128: 1024 x 16B chunks ----
        #pragma unroll
        for (int i = 0; i < 4; ++i) {
            int c  = tid + i * 256;        // consecutive tids -> consecutive 16B
            int r  = c >> 3;               // row 0..127
            int k8 = (c & 7) << 3;         // 0,8,...,56 halves
            async_ld_b128(asBase + (unsigned)(r * LDA + k8) * 2,
                          A + (size_t)(mBlk + r) * K + kb + k8);
        }
        asm volatile("s_wait_asynccnt 0" ::: "memory");
        if (w == 0)
            __builtin_amdgcn_s_wait_tensorcnt(0);
        __syncthreads();

        #pragma unroll
        for (int ks = 0; ks < BK; ks += 32) {
            v16bf a[2], b[2];
            #pragma unroll
            for (int tm = 0; tm < 2; ++tm)
                a[tm] = ld_frag(&As[(wm * 32 + tm * 16 + l15) * LDA + ks + hsel]);
            #pragma unroll
            for (int tn = 0; tn < 2; ++tn)
                b[tn] = ld_frag(&Bs[(wn * 32 + tn * 16 + l15) * LDB + ks + hsel]);
            #pragma unroll
            for (int tm = 0; tm < 2; ++tm)
                #pragma unroll
                for (int tn = 0; tn < 2; ++tn)
                    acc[tm][tn] = __builtin_amdgcn_wmma_f32_16x16x32_bf16(
                        false, a[tm], false, b[tn], (short)0, acc[tm][tn], false, false);
        }
        __syncthreads();
    }

    // ---- epilogue: C 16x16 f32 layout (VGPR r -> M=r / r+8 by lane half) ----
    const int mo = (lane >> 4) * 8;
    #pragma unroll
    for (int tm = 0; tm < 2; ++tm) {
        #pragma unroll
        for (int tn = 0; tn < 2; ++tn) {
            int n = nBlk + wn * 32 + tn * 16 + l15;
            float bn = bias ? bias[n] : 0.0f;
            #pragma unroll
            for (int r = 0; r < 8; ++r) {
                int m = mBlk + wm * 32 + tm * 16 + mo + r;
                float v = acc[tm][tn][r] + bn;
                if (rowBias) v += rowBias[(size_t)(m / rowDiv) * N + n];
                if (doRelu)  v = fmaxf(v, 0.0f);
                size_t idx = (size_t)m * N + n;
                if (Cf) Cf[idx] = v;
                if (Cb) Cb[idx] = f2bf(v);
            }
        }
    }
}

// ---------------------------------------------------------------------------
// Weight prep: Wt[n*K+k] = bf16(W[k*N+n]) for a 512x512 fp32 matrix.
// 32x32 LDS tile transpose, grid (16,16), 256 threads.
// ---------------------------------------------------------------------------
__global__ __launch_bounds__(256)
void wtrans_kernel(const float* __restrict__ W, bf16_t* __restrict__ Wt)
{
    __shared__ float tile[32][33];
    int bx = blockIdx.x * 32, by = blockIdx.y * 32;   // bx: n, by: k
    int tx = threadIdx.x & 31, ty4 = (threadIdx.x >> 5) * 4;
    #pragma unroll
    for (int i = 0; i < 4; ++i)
        tile[ty4 + i][tx] = W[(size_t)(by + ty4 + i) * H_ + bx + tx];
    __syncthreads();
    #pragma unroll
    for (int i = 0; i < 4; ++i)
        Wt[(size_t)(bx + ty4 + i) * H_ + by + tx] = f2bf(tile[tx][ty4 + i]);
}

// ---------------------------------------------------------------------------
// Embedding gather: x[b,t,:] = embed[ids[b,t],:]  (float4 vectorized)
// ---------------------------------------------------------------------------
__global__ __launch_bounds__(256)
void embed_kernel(const int* __restrict__ ids, const float* __restrict__ embed,
                  float* __restrict__ x)
{
    size_t idx = (size_t)blockIdx.x * 256 + threadIdx.x;
    size_t bt  = idx >> 7;
    int    h4  = (int)(idx & 127) << 2;
    int    id  = ids[bt];
    float4 v = *(const float4*)(embed + (size_t)id * H_ + h4);
    *(float4*)(x + bt * H_ + h4) = v;
}

// ---------------------------------------------------------------------------
// Timestep embedder: temb = [cos(t*f), sin(t*f)]; c = gelu(temb@w1+b1)@w2+b2
// ---------------------------------------------------------------------------
__global__ __launch_bounds__(128)
void timestep_kernel(const float* __restrict__ t,
                     const float* __restrict__ w1, const float* __restrict__ b1,
                     const float* __restrict__ w2, const float* __restrict__ b2,
                     float* __restrict__ c)
{
    __shared__ float temb[FE_];
    __shared__ float h1[CD_];
    int b = blockIdx.x, tid = threadIdx.x;
    float tv = t[b];
    float fr = __expf(-9.210340371976184f * (float)tid / 128.0f);
    float a  = tv * fr;
    temb[tid]       = cosf(a);
    temb[tid + 128] = sinf(a);
    __syncthreads();
    float y = b1[tid];
    for (int i = 0; i < FE_; ++i) y += temb[i] * w1[(size_t)i * CD_ + tid];
    h1[tid] = 0.5f * y * (1.0f + erff(y * 0.7071067811865476f));  // exact GELU
    __syncthreads();
    float z = b2[tid];
    for (int i = 0; i < CD_; ++i) z += h1[i] * w2[(size_t)i * CD_ + tid];
    c[(size_t)b * CD_ + tid] = z;
}

// cproj[b,n] = out_b[n] + sum_i c[b,i] * out_w[(H+i), n]
__global__ __launch_bounds__(256)
void cproj_kernel(const float* __restrict__ c, const float* __restrict__ out_w,
                  const float* __restrict__ out_b, float* __restrict__ cproj)
{
    __shared__ float cs[CD_];
    int b = blockIdx.x, tid = threadIdx.x;
    if (tid < CD_) cs[tid] = c[(size_t)b * CD_ + tid];
    __syncthreads();
    #pragma unroll
    for (int s = 0; s < 2; ++s) {
        int n = tid + s * 256;
        float acc = out_b[n];
        for (int i = 0; i < CD_; ++i)
            acc += cs[i] * out_w[(size_t)(H_ + i) * H_ + n];
        cproj[(size_t)b * H_ + n] = acc;
    }
}

// ---------------------------------------------------------------------------
// block reduction helper (wave32 shfl + 8-wave shared combine)
// ---------------------------------------------------------------------------
__device__ __forceinline__ float block_sum(float v, float* sh) {
    int lane = threadIdx.x & 31, w = threadIdx.x >> 5;
    #pragma unroll
    for (int o = 16; o; o >>= 1) v += __shfl_xor(v, o, 32);
    if (lane == 0) sh[w] = v;
    __syncthreads();
    float r = (lane < 8) ? sh[lane] : 0.0f;
    if (w == 0) {
        #pragma unroll
        for (int o = 4; o; o >>= 1) r += __shfl_xor(r, o, 32);
        if (lane == 0) sh[0] = r;
    }
    __syncthreads();
    float out = sh[0];
    __syncthreads();
    return out;
}

// ---------------------------------------------------------------------------
// Rotary + kmask + norm1 + qmask. e emitted bf16 (GEMM input only);
// qn emitted both f32 (attention residual) and bf16 (Q GEMM input).
// ---------------------------------------------------------------------------
__global__ __launch_bounds__(256)
void rope_norm1_kernel(const float* __restrict__ x, const float* __restrict__ n1w,
                       bf16_t* __restrict__ ebf, float* __restrict__ qn,
                       bf16_t* __restrict__ qnbf,
                       float* __restrict__ kmask, float* __restrict__ qmask)
{
    __shared__ float sh[8];
    size_t tok  = blockIdx.x;
    int    tpos = (int)(tok % T_);
    int    tid  = threadIdx.x;
    const float* xi = x + tok * H_;

    float ev[2];
    #pragma unroll
    for (int s = 0; s < 2; ++s) {
        int dd = tid + s * 256;
        int cc = dd & 255;
        float th  = __expf(-9.210340371976184f * (2.0f * (float)cc) / (float)H_);
        float ang = (float)tpos * th;
        float rh  = (dd < 256) ? -xi[dd + 256] : xi[dd - 256];
        ev[s] = xi[dd] * cosf(ang) + rh * sinf(ang);
    }
    float se  = block_sum(ev[0] + ev[1], sh);
    float sq  = block_sum(ev[0] * ev[0] + ev[1] * ev[1], sh);
    float mean = se / (float)H_;
    float var  = sq / (float)H_ - mean * mean;
    float rstd = rsqrtf(var + 1e-5f);

    float qv[2];
    #pragma unroll
    for (int s = 0; s < 2; ++s) {
        int dd = tid + s * 256;
        ebf[tok * H_ + dd] = f2bf(ev[s]);
        qv[s] = (ev[s] - mean) * rstd * n1w[dd];
        qn[tok * H_ + dd]   = qv[s];
        qnbf[tok * H_ + dd] = f2bf(qv[s]);
    }
    float sqn = block_sum(qv[0] + qv[1], sh);
    if (tid == 0) {
        kmask[tok] = (fabsf(se)  > 0.0f) ? 1.0f : 0.0f;
        qmask[tok] = (fabsf(sqn) > 0.0f) ? 1.0f : 0.0f;
    }
}

// ---------------------------------------------------------------------------
// LayerNorm: out = LN(in (+ add)) * w (+ bias); optional bf16 mirror output.
// ---------------------------------------------------------------------------
__global__ __launch_bounds__(256)
void ln_kernel(const float* __restrict__ in, const float* __restrict__ add,
               const float* __restrict__ w, const float* __restrict__ bias,
               float* __restrict__ out, bf16_t* __restrict__ outb)
{
    __shared__ float sh[8];
    size_t tok = blockIdx.x;
    int tid = threadIdx.x;
    float v[2];
    #pragma unroll
    for (int s = 0; s < 2; ++s) {
        int dd = tid + s * 256;
        v[s] = in[tok * H_ + dd] + (add ? add[tok * H_ + dd] : 0.0f);
    }
    float se = block_sum(v[0] + v[1], sh);
    float sq = block_sum(v[0] * v[0] + v[1] * v[1], sh);
    float mean = se / (float)H_;
    float var  = sq / (float)H_ - mean * mean;
    float rstd = rsqrtf(var + 1e-5f);
    #pragma unroll
    for (int s = 0; s < 2; ++s) {
        int dd = tid + s * 256;
        float y = (v[s] - mean) * rstd * w[dd] + (bias ? bias[dd] : 0.0f);
        if (out)  out[tok * H_ + dd]  = y;
        if (outb) outb[tok * H_ + dd] = f2bf(y);
    }
}

// ---------------------------------------------------------------------------
// Attention. One block per (b, head): K staged in LDS (padded stride 66),
// one wave per query row, wave32 softmax, fused att = o + qn residual.
// ---------------------------------------------------------------------------
__global__ __launch_bounds__(256)
void attention_kernel(const float* __restrict__ Q, const float* __restrict__ Kg,
                      const float* __restrict__ Vg, const float* __restrict__ kmask,
                      const float* __restrict__ qmask, const float* __restrict__ qn,
                      float* __restrict__ att)
{
    __shared__ float Ks[T_ * 66];
    __shared__ float qall[8 * 66];
    const int bh = blockIdx.x;
    const int b  = bh >> 3;
    const int h  = bh & 7;
    const size_t baseBT = (size_t)b * T_;
    const int tid = threadIdx.x, lane = tid & 31, w = tid >> 5;

    for (int idx = tid; idx < T_ * DH_; idx += 256) {
        int j = idx >> 6, d = idx & 63;
        Ks[j * 66 + d] = Kg[(baseBT + j) * H_ + h * DH_ + d];
    }
    __syncthreads();

    const float scale = 0.044194173824159216f;   // 1/sqrt(512)
    const float NEGV  = -4294967295.0f;          // -2^32+1

    for (int it = 0; it < T_ / 8; ++it) {
        #pragma unroll
        for (int s = 0; s < 2; ++s) {
            int idx = tid + s * 256;             // 512 = 8*64
            int qi = idx >> 6, d = idx & 63;
            qall[qi * 66 + d] = Q[(baseBT + it * 8 + qi) * H_ + h * DH_ + d];
        }
        __syncthreads();

        const int i = it * 8 + w;
        float sc[7];
        float mx = -3.0e38f;
        #pragma unroll
        for (int jt = 0; jt < 7; ++jt) {
            int j = jt * 32 + lane;
            float s;
            if (j >= T_) {
                s = -INFINITY;                   // phantom key: excluded
            } else if (j <= i && kmask[baseBT + j] != 0.0f) {
                float acc = 0.0f;
                #pragma unroll 8
                for (int d = 0; d < DH_; ++d)
                    acc += qall[w * 66 + d] * Ks[j * 66 + d];
                s = acc * scale;
            } else {
                s = NEGV;
            }
            sc[jt] = s;
            mx = fmaxf(mx, s);
        }
        #pragma unroll
        for (int o = 16; o; o >>= 1) mx = fmaxf(mx, __shfl_xor(mx, o, 32));
        float sum = 0.0f;
        #pragma unroll
        for (int jt = 0; jt < 7; ++jt) {
            float p = __expf(sc[jt] - mx);
            sc[jt] = p; sum += p;
        }
        #pragma unroll
        for (int o = 16; o; o >>= 1) sum += __shfl_xor(sum, o, 32);
        float inv = qmask[baseBT + i] / sum;
        #pragma unroll
        for (int jt = 0; jt < 7; ++jt) sc[jt] *= inv;

        const int d0 = lane * 2;
        float o0 = 0.0f, o1 = 0.0f;
        for (int jt = 0; jt < 7; ++jt) {
            float pl = sc[jt];
            for (int lj = 0; lj < 32; ++lj) {
                int j = jt * 32 + lj;
                if (j >= T_) break;              // uniform across wave
                float p = __shfl(pl, lj, 32);
                const float* vrow = Vg + (baseBT + j) * H_ + h * DH_;
                o0 += p * vrow[d0];
                o1 += p * vrow[d0 + 1];
            }
        }
        size_t oi = (baseBT + i) * H_ + h * DH_ + d0;
        att[oi]     = o0 + qn[oi];
        att[oi + 1] = o1 + qn[oi + 1];
        __syncthreads();
    }
}

// ---------------------------------------------------------------------------
extern "C" void kernel_launch(void* const* d_in, const int* in_sizes, int n_in,
                              void* d_out, int out_size, void* d_ws, size_t ws_size,
                              hipStream_t stream) {
    (void)in_sizes; (void)n_in; (void)out_size; (void)ws_size;

    const int*   ids   = (const int*)  d_in[0];
    const float* t     = (const float*)d_in[1];
    const float* embed = (const float*)d_in[2];
    const float* tw_w1 = (const float*)d_in[3];
    const float* tw_b1 = (const float*)d_in[4];
    const float* tw_w2 = (const float*)d_in[5];
    const float* tw_b2 = (const float*)d_in[6];
    const float* n1    = (const float*)d_in[7];
    const float* wq    = (const float*)d_in[8];
    const float* bq    = (const float*)d_in[9];
    const float* wk    = (const float*)d_in[10];
    const float* bk    = (const float*)d_in[11];
    const float* wv    = (const float*)d_in[12];
    const float* bv    = (const float*)d_in[13];
    const float* n2    = (const float*)d_in[14];
    const float* fw1   = (const float*)d_in[15];
    const float* fb1   = (const float*)d_in[16];
    const float* fw2   = (const float*)d_in[17];
    const float* fb2   = (const float*)d_in[18];
    const float* lnw   = (const float*)d_in[19];
    const float* lnb   = (const float*)d_in[20];
    const float* n3    = (const float*)d_in[21];
    const float* out_w = (const float*)d_in[22];
    const float* out_b = (const float*)d_in[23];

    float* out = (float*)d_out;
    float* ws  = (float*)d_ws;

    const size_t BTH = (size_t)B_ * T_ * H_;          // 26,214,400
    const size_t BT  = (size_t)B_ * T_;               // 51,200
    float* W_E  = ws;                                 // att / pre-norm buffers
    float* W_QN = ws + 1 * BTH;                       // qn / h2 (f32)
    float* W_Q  = ws + 2 * BTH;                       // Q (f32)
    float* W_K  = ws + 3 * BTH;                       // K / ff2 (f32)
    float* W_X  = ws + 4 * BTH;                       // x (f32)
    float* W_V  = out;                                // V scratch in d_out
    float* W_KM = ws + 5 * BTH;                       // kmask [B*T]
    float* W_QM = W_KM + 65536;                       // qmask [B*T]
    float* W_C  = W_QM + 65536;                       // c     [B,CD]
    float* W_CP = W_C  + (size_t)B_ * CD_;            // cproj [B,H]
    bf16_t* ABF0 = (bf16_t*)(W_CP + (size_t)B_ * H_); // bf16 activations 0
    bf16_t* ABF1 = ABF0 + BTH;                        // bf16 activations 1
    bf16_t* WB   = ABF1 + BTH;                        // bf16 [N,K] weights x11

    const int M = (int)BT, N = H_, K = H_;
    dim3 ggrid(M / BM, N / BN);                       // (400, 8)
    dim3 tgrid(16, 16);

    // 0) weight prep: transpose + bf16 (11 matrices, ~6 MB total)
    for (int l = 0; l < L_; ++l) {
        const size_t wo = (size_t)l * HH_;
        wtrans_kernel<<<tgrid, 256, 0, stream>>>(wq  + wo, WB + (size_t)(l * 5 + 0) * HH_);
        wtrans_kernel<<<tgrid, 256, 0, stream>>>(wk  + wo, WB + (size_t)(l * 5 + 1) * HH_);
        wtrans_kernel<<<tgrid, 256, 0, stream>>>(wv  + wo, WB + (size_t)(l * 5 + 2) * HH_);
        wtrans_kernel<<<tgrid, 256, 0, stream>>>(fw1 + wo, WB + (size_t)(l * 5 + 3) * HH_);
        wtrans_kernel<<<tgrid, 256, 0, stream>>>(fw2 + wo, WB + (size_t)(l * 5 + 4) * HH_);
    }
    wtrans_kernel<<<tgrid, 256, 0, stream>>>(out_w, WB + (size_t)10 * HH_);

    // 1) x = embed[ids] ; 2) timestep MLP ; 3) cproj = out_b + c @ out_w[H:,:]
    embed_kernel<<<(unsigned)(BTH / 4 / 256), 256, 0, stream>>>(ids, embed, W_X);
    timestep_kernel<<<B_, 128, 0, stream>>>(t, tw_w1, tw_b1, tw_w2, tw_b2, W_C);
    cproj_kernel<<<B_, 256, 0, stream>>>(W_C, out_w, out_b, W_CP);

    for (int l = 0; l < L_; ++l) {
        const size_t bo = (size_t)l * H_;
        const bf16_t* Wq = WB + (size_t)(l * 5 + 0) * HH_;
        const bf16_t* Wk = WB + (size_t)(l * 5 + 1) * HH_;
        const bf16_t* Wv = WB + (size_t)(l * 5 + 2) * HH_;
        const bf16_t* W1 = WB + (size_t)(l * 5 + 3) * HH_;
        const bf16_t* W2 = WB + (size_t)(l * 5 + 4) * HH_;

        // rotary + kmask + norm1 + qmask (e -> ABF0, qn -> f32 + ABF1)
        rope_norm1_kernel<<<(unsigned)BT, 256, 0, stream>>>(
            W_X, n1 + bo, ABF0, W_QN, ABF1, W_KM, W_QM);
        // Q = qn @ wq + bq ; K = e @ wk + bk ; V = e @ wv + bv   (WMMA)
        gemm_bf16_wmma<<<ggrid, 256, 0, stream>>>(ABF1, Wq, bq + bo, nullptr,
                                                  W_Q, nullptr, M, N, K, 1, 0);
        gemm_bf16_wmma<<<ggrid, 256, 0, stream>>>(ABF0, Wk, bk + bo, nullptr,
                                                  W_K, nullptr, M, N, K, 1, 0);
        gemm_bf16_wmma<<<ggrid, 256, 0, stream>>>(ABF0, Wv, bv + bo, nullptr,
                                                  W_V, nullptr, M, N, K, 1, 0);
        // att = softmax(QK^T/sqrt(H), masks) @ V + qn
        attention_kernel<<<B_ * NH_, 256, 0, stream>>>(
            W_Q, W_K, W_V, W_KM, W_QM, W_QN, W_E);
        // h2 = LN(att) * norm2  (f32 + bf16)
        ln_kernel<<<(unsigned)BT, 256, 0, stream>>>(W_E, nullptr, n2 + bo, nullptr,
                                                    W_QN, ABF0);
        // ff1 = relu(h2 @ ffw1 + ffb1)  (bf16 only) ; ff2 = ff1 @ ffw2 + ffb2
        gemm_bf16_wmma<<<ggrid, 256, 0, stream>>>(ABF0, W1, fb1 + bo, nullptr,
                                                  nullptr, ABF1, M, N, K, 1, 1);
        gemm_bf16_wmma<<<ggrid, 256, 0, stream>>>(ABF1, W2, fb2 + bo, nullptr,
                                                  W_K, nullptr, M, N, K, 1, 0);
        // ff = LN(ff2 + h2) * lnw + lnb ; x = LN(ff) * norm3 (f32 + bf16)
        ln_kernel<<<(unsigned)BT, 256, 0, stream>>>(W_K, W_QN, lnw + bo, lnb + bo,
                                                    W_E, nullptr);
        ln_kernel<<<(unsigned)BT, 256, 0, stream>>>(W_E, nullptr, n3 + bo, nullptr,
                                                    W_X, ABF0);
    }

    // 5) out = x @ out_w[:H,:] + cproj[b,:]   (per-batch-row bias)
    gemm_bf16_wmma<<<ggrid, 256, 0, stream>>>(ABF0, WB + (size_t)10 * HH_,
                                              nullptr, W_CP, out, nullptr,
                                              M, N, K, T_, 0);
}
